// EfficientMixedScoreMultiHeadAttention_62629213110578
// MI455X (gfx1250) — matrix operand order repaired and checked by reading, not compiled, not run on hardware
//
#include <hip/hip_runtime.h>
#include <hip/hip_bf16.h>
#include <math.h>
#include <stdint.h>

typedef __attribute__((ext_vector_type(16))) _Float16 v16h;
typedef __attribute__((ext_vector_type(8)))  float    v8f;

#define B_    8
#define R_    128
#define C_    128
#define E_    256
#define H_    16
#define D_    16
#define HID_  512      /* H*MS */
#define F2H_  32       /* 2*H  */
#define QVW_  512      /* 2*E  */

// ---------- WMMA fragment helpers (wave32, v_wmma_f32_16x16x32_f16) ----------

// A-matrix 16x32 f16: lane holds row M=lane%16; half e -> K = e + 16g.
__device__ __forceinline__ int a_k(int e, int g) {
    return e + (g << 4);
}

__device__ __forceinline__ v16h load_a_rm(const _Float16* __restrict__ A, int ld, int row0, int k0) {
    int lane = threadIdx.x & 31;
    int m = lane & 15, g = lane >> 4;
    v16h a;
#pragma unroll
    for (int e = 0; e < 16; ++e)
        a[e] = A[(row0 + m) * ld + k0 + a_k(e, g)];
    return a;
}

// Packed B fragment: P[((kt*ntiles + nt)*32 + lane)*16 + e]  (32B/lane, aligned)
__device__ __forceinline__ v16h load_b_pk(const _Float16* __restrict__ P, int ntiles, int kt, int nt) {
    int lane = threadIdx.x & 31;
    return *(const v16h*)(P + (((size_t)kt * ntiles + nt) * 32 + lane) * 16);
}

__device__ __forceinline__ v8f wmma16(v16h a, v16h b, v8f c) {
    return __builtin_amdgcn_wmma_f32_16x16x32_f16(false, a, false, b, (short)0, c, false, false);
}

// ---------- elementwise ----------

__global__ void k_f2h(const float* __restrict__ s, _Float16* __restrict__ d, int n) {
    int i = blockIdx.x * blockDim.x + threadIdx.x;
    if (i < n) d[i] = (_Float16)s[i];
}

__global__ void k_zero(float* __restrict__ p, int n) {
    int i = blockIdx.x * blockDim.x + threadIdx.x;
    if (i < n) p[i] = 0.0f;
}

// ---------- pack a row-major f32 weight [KxN] into f16 B-fragment-major order ----------
// B layout: lane holds col N=lane%16, g=lane/16; half e -> K = k0 + e + 16g.
__global__ void __launch_bounds__(128)
k_pack_b(const float* __restrict__ W, _Float16* __restrict__ P, int K, int N) {
    int wave = (int)((blockIdx.x * blockDim.x + threadIdx.x) >> 5);
    int ntiles = N >> 4, ktiles = K >> 5;
    if (wave >= ktiles * ntiles) return;
    int kt = wave / ntiles, nt = wave % ntiles;
    int lane = threadIdx.x & 31;
    int n = lane & 15, g = lane >> 4;
    _Float16* dst = P + (((size_t)kt * ntiles + nt) * 32 + lane) * 16;
#pragma unroll
    for (int e = 0; e < 16; ++e)
        dst[e] = (_Float16)W[((kt << 5) + e + (g << 4)) * N + (nt << 4) + n];
}

// ---------- pack v-part of qvh into B-fragment order for attention combine ----------
// vp[(((b*H+h)*4 + kt)*32 + lane)*16 + e] = v[b,h, t=kt*32+e+16g, d=lane%16]
__global__ void __launch_bounds__(128)
k_packv(const _Float16* __restrict__ qvh, _Float16* __restrict__ vp) {
    int wave = (int)((blockIdx.x * blockDim.x + threadIdx.x) >> 5);   // 512 waves
    if (wave >= 512) return;
    int kt = wave & 3; wave >>= 2;
    int h  = wave & 15; wave >>= 4;
    int b  = wave;
    int lane = threadIdx.x & 31;
    int n = lane & 15, g = lane >> 4;
    _Float16* dst = vp + ((size_t)((b * H_ + h) * 4 + kt) * 32 + lane) * 16;
#pragma unroll
    for (int e = 0; e < 16; ++e) {
        int t = (kt << 5) + e + (g << 4);
        dst[e] = qvh[(b * 128 + t) * QVW_ + E_ + h * D_ + n];
    }
}

// ---------- generic f16 GEMM: C[MxN] = A[MxK] @ Bpacked[KxN] ----------
// STORE_F32 selects the output type at compile time -> branch-free epilogue.
template <bool STORE_F32>
__global__ void __launch_bounds__(128)
k_gemm_f16(const _Float16* __restrict__ A, const _Float16* __restrict__ Bp,
           float* __restrict__ Cf, _Float16* __restrict__ Ch, int M, int N, int K) {
    int wave = (int)((blockIdx.x * blockDim.x + threadIdx.x) >> 5);
    int ntile = N >> 4;
    int tiles = (M >> 4) * ntile;
    if (wave >= tiles) return;                      // whole-wave uniform: EXEC stays all-1s
    int mt = wave / ntile, nt = wave % ntile;
    v8f acc = {};
    for (int k0 = 0; k0 < K; k0 += 32) {
        __builtin_prefetch(&A[(mt << 4) * K + k0 + 32], 0, 0);       // global_prefetch_b8
        v16h a = load_a_rm(A, K, mt << 4, k0);
        v16h b = load_b_pk(Bp, ntile, k0 >> 5, nt);
        acc = wmma16(a, b, acc);
    }
    int lane = threadIdx.x & 31;
    int n = (nt << 4) + (lane & 15);
    int g = lane >> 4;
#pragma unroll
    for (int i = 0; i < 8; ++i) {
        int mr = (mt << 4) + i + (g << 3);
        if (STORE_F32) Cf[mr * N + n] = acc[i];
        else           Ch[mr * N + n] = (_Float16)acc[i];
    }
}

// ---------- dot = 0.25 * q @ k^T, per (b,h), K=16 zero-padded to 32 ----------
__global__ void __launch_bounds__(128)
k_dot(const _Float16* __restrict__ qvh1, const _Float16* __restrict__ qvh2,
      float* __restrict__ dotb) {
    int wave = (int)((blockIdx.x * blockDim.x + threadIdx.x) >> 5);   // 8192 waves
    int ct = wave & 7; wave >>= 3;
    int rt = wave & 7; wave >>= 3;
    int h  = wave & 15; wave >>= 4;
    int b  = wave;
    int lane = threadIdx.x & 31;
    int m = lane & 15, g = lane >> 4;

    v16h a, bb;
#pragma unroll
    for (int e = 0; e < 16; ++e) {
        int ka = a_k(e, g);
        int r  = (rt << 4) + m;
        a[e] = (ka < 16) ? qvh1[(b * R_ + r) * QVW_ + h * D_ + ka] : (_Float16)0.0f;
        int kb = e + (g << 4);
        int c  = (ct << 4) + m;
        bb[e] = (kb < 16) ? qvh2[(b * C_ + c) * QVW_ + h * D_ + kb] : (_Float16)0.0f;
    }
    v8f acc = {};
    acc = wmma16(a, bb, acc);
#pragma unroll
    for (int i = 0; i < 8; ++i) {
        int r = (rt << 4) + i + (g << 3);
        int c = (ct << 4) + (lane & 15);
        dotb[((b * H_ + h) * R_ + r) * C_ + c] = acc[i] * 0.25f;     // NORM = 1/sqrt(16)
    }
}

// ---------- fused MixedScoreFF MLP ----------
// feat[16pos x 32] -> relu(@lin1[32x512]) -> @lin2[512x32]; lin2 staged to LDS with
// global_load_async_to_lds_b128 (ASYNCcnt), overlapped with layer-1 WMMAs.
__global__ void __launch_bounds__(64)
k_mlp(const float* __restrict__ dotb, const float* __restrict__ cost,
      const _Float16* __restrict__ l1p, const _Float16* __restrict__ l2p,
      float* __restrict__ ms1, float* __restrict__ ms2, float* __restrict__ acc4) {
    __shared__ _Float16 hid[2][16][HID_];   // 32 KB
    __shared__ _Float16 l2s[16384];         // 32 KB: packed lin2 (16 ktiles x 2 ntiles)

    // ---- issue async staging of packed lin2 into LDS (16B per lane per op) ----
    {
        uint32_t ldsbase = (uint32_t)(uintptr_t)&l2s[0];   // low 32 bits = LDS offset
        const char* gsrc = (const char*)l2p;
#pragma unroll 4
        for (int it = 0; it < 32; ++it) {
            uint32_t off = (uint32_t)((it * 64 + threadIdx.x) * 16);
            uint32_t la = ldsbase + off;
            unsigned long long ga = (unsigned long long)(uintptr_t)(gsrc + off);
            asm volatile("global_load_async_to_lds_b128 %0, %1, off"
                         :: "v"(la), "v"(ga) : "memory");
        }
    }

    int wv   = threadIdx.x >> 5;
    int wave = (int)((blockIdx.x * blockDim.x + threadIdx.x) >> 5);   // 8192 waves
    int ct = wave & 7;  wave >>= 3;
    int r  = wave & 127; wave >>= 7;
    int b  = wave;
    int lane = threadIdx.x & 31;
    int m = lane & 15, g = lane >> 4;

    // A fragment: interleaved features feat[pos=c][2h]=dot, [2h+1]=cost
    v16h af;
#pragma unroll
    for (int e = 0; e < 16; ++e) {
        int k = a_k(e, g);
        int c = (ct << 4) + m;
        int h = k >> 1;
        float v = (k & 1) ? cost[(b * R_ + r) * C_ + c]
                          : dotb[((b * H_ + h) * R_ + r) * C_ + c];
        af[e] = (_Float16)v;
    }

    // layer 1: K=32 exactly -> one WMMA per 16-wide N tile; relu -> LDS
#pragma unroll 4
    for (int nt = 0; nt < 32; ++nt) {
        v16h bf = load_b_pk(l1p, 32, 0, nt);
        v8f c0 = {};
        c0 = wmma16(af, bf, c0);
#pragma unroll
        for (int i = 0; i < 8; ++i) {
            float v = c0[i];
            hid[wv][i + (g << 3)][(nt << 4) + (lane & 15)] = (_Float16)(v > 0.0f ? v : 0.0f);
        }
    }

    // staged weights must be complete (and visible across both waves) before layer 2
    asm volatile("s_wait_asynccnt 0x0" ::: "memory");
    __syncthreads();   // safe: exactly one tile per wave, uniform control flow

    // layer 2: K=512 in 16 steps, N=32 in two tiles; B from LDS (ds_load_b128)
    v8f a0 = {}, a1 = {};
    for (int kt = 0; kt < 16; ++kt) {
        v16h ah;
#pragma unroll
        for (int e = 0; e < 16; ++e)
            ah[e] = hid[wv][m][(kt << 5) + a_k(e, g)];
        v16h b0 = *(const v16h*)&l2s[(((kt << 1) + 0) * 32 + lane) * 16];
        v16h b1 = *(const v16h*)&l2s[(((kt << 1) + 1) * 32 + lane) * 16];
        a0 = wmma16(ah, b0, a0);
        a1 = wmma16(ah, b1, a1);
    }

    // split ms1 (even feature) / ms2 (odd feature); accumulate stats
    int n = lane & 15;
    float s = 0.0f, sq = 0.0f;
#pragma unroll
    for (int i = 0; i < 8; ++i) {
        int c = (ct << 4) + i + (g << 3);
        {
            int j = n; float v = a0[i];
            float* dst = (j & 1) ? ms2 : ms1;
            dst[((b * H_ + (j >> 1)) * R_ + r) * C_ + c] = v;
            s += v; sq += v * v;
        }
        {
            int j = 16 + n; float v = a1[i];
            float* dst = (j & 1) ? ms2 : ms1;
            dst[((b * H_ + (j >> 1)) * R_ + r) * C_ + c] = v;
            s += v; sq += v * v;
        }
    }
    float s1 = (n & 1) ? 0.0f : s,  q1 = (n & 1) ? 0.0f : sq;
    float s2 = (n & 1) ? s : 0.0f,  q2 = (n & 1) ? sq : 0.0f;
#pragma unroll
    for (int off = 16; off > 0; off >>= 1) {
        s1 += __shfl_down(s1, off, 32);
        q1 += __shfl_down(q1, off, 32);
        s2 += __shfl_down(s2, off, 32);
        q2 += __shfl_down(q2, off, 32);
    }
    if (lane == 0) {
        atomicAdd(&acc4[0], s1); atomicAdd(&acc4[1], q1);
        atomicAdd(&acc4[2], s2); atomicAdd(&acc4[3], q2);
    }
}

// ---------- tiny: unbiased std -> inverse ----------
__global__ void k_std(float* __restrict__ acc4) {
    const float n = (float)(B_ * H_ * R_ * C_);   // 2097152 per tensor
    float v1 = (acc4[1] - acc4[0] * acc4[0] / n) / (n - 1.0f);
    float v2 = (acc4[3] - acc4[2] * acc4[2] / n) / (n - 1.0f);
    acc4[4] = rsqrtf(fmaxf(v1, 1e-30f));
    acc4[5] = rsqrtf(fmaxf(v2, 1e-30f));
}

// ---------- tanh/softmax + w@v.  transposed=1 reads ms[b,h,t,s] (swapaxes). ----------
__global__ void __launch_bounds__(128)
k_att(const float* __restrict__ ms, const _Float16* __restrict__ vpk,
      const float* __restrict__ acc4, int statIdx, int transposed,
      _Float16* __restrict__ outh) {
    __shared__ _Float16 wbuf[4][16][128];                             // 16 KB
    int wv   = threadIdx.x >> 5;
    int wave = (int)((blockIdx.x * blockDim.x + threadIdx.x) >> 5);   // 1024 waves
    int st = wave & 7;  wave >>= 3;
    int h  = wave & 15; wave >>= 4;
    int b  = wave;
    int lane = threadIdx.x & 31;
    float inv = acc4[statIdx];

    for (int rr = 0; rr < 16; ++rr) {
        int s = (st << 4) + rr;
        float l[4];
        float mx = -1e30f;
#pragma unroll
        for (int j = 0; j < 4; ++j) {
            int t = lane + (j << 5);
            float v = transposed ? ms[((b * H_ + h) * R_ + t) * C_ + s]
                                 : ms[((b * H_ + h) * R_ + s) * C_ + t];
            v = tanhf(v * inv) * 10.0f;
            l[j] = v;
            mx = fmaxf(mx, v);
        }
#pragma unroll
        for (int off = 16; off > 0; off >>= 1) mx = fmaxf(mx, __shfl_xor(mx, off, 32));
        float sum = 0.0f;
#pragma unroll
        for (int j = 0; j < 4; ++j) { l[j] = __expf(l[j] - mx); sum += l[j]; }
#pragma unroll
        for (int off = 16; off > 0; off >>= 1) sum += __shfl_xor(sum, off, 32);
        float rs = (sum > 0.0f) ? (1.0f / sum) : 0.0f;
#pragma unroll
        for (int j = 0; j < 4; ++j) {
            float w = l[j] * rs;
            if (!(w == w)) w = 0.0f;                                   // nan_to_num
            wbuf[wv][rr][lane + (j << 5)] = (_Float16)w;
        }
    }
    __syncthreads();

    // out[16s x 16d] = w[16 x 128] @ v[128 x 16]; V from packed fragments
    int m = lane & 15, g = lane >> 4;
    v8f acc = {};
    for (int kt = 0; kt < 4; ++kt) {
        v16h a;
#pragma unroll
        for (int e = 0; e < 16; ++e) a[e] = wbuf[wv][m][(kt << 5) + a_k(e, g)];
        v16h bf = *(const v16h*)(vpk + ((size_t)((b * H_ + h) * 4 + kt) * 32 + lane) * 16);
        acc = wmma16(a, bf, acc);
    }
#pragma unroll
    for (int i = 0; i < 8; ++i) {
        int s = (st << 4) + i + (g << 3);
        outh[(b * 128 + s) * E_ + h * D_ + (lane & 15)] = (_Float16)acc[i];
    }
}

// ---------- host ----------

static inline char* take(char*& p, size_t bytes) {
    char* r = p;
    p += (bytes + 255) & ~(size_t)255;
    return r;
}

extern "C" void kernel_launch(void* const* d_in, const int* in_sizes, int n_in,
                              void* d_out, int out_size, void* d_ws, size_t ws_size,
                              hipStream_t stream) {
    (void)in_sizes; (void)n_in; (void)out_size; (void)ws_size;
    const float* x1   = (const float*)d_in[0];   // [8,128,256]
    const float* x2   = (const float*)d_in[1];   // [8,128,256]
    const float* cost = (const float*)d_in[2];   // [8,128,128]
    const float* Wqv1 = (const float*)d_in[3];   // [256,512]
    const float* lin1 = (const float*)d_in[4];   // [32,512]
    const float* lin2 = (const float*)d_in[5];   // [512,32]
    const float* out1 = (const float*)d_in[6];   // [256,256]
    const float* out2 = (const float*)d_in[7];   // [256,256]
    float* dOut = (float*)d_out;                 // h1 [8,128,256] then h2 [8,128,256]

    char* p = (char*)d_ws;
    _Float16* x1h   = (_Float16*)take(p, 262144 * 2);
    _Float16* x2h   = (_Float16*)take(p, 262144 * 2);
    _Float16* Wqp   = (_Float16*)take(p, 131072 * 2);   // packed B frags
    _Float16* l1p   = (_Float16*)take(p, 16384 * 2);
    _Float16* l2p   = (_Float16*)take(p, 16384 * 2);
    _Float16* o1p   = (_Float16*)take(p, 65536 * 2);
    _Float16* o2p   = (_Float16*)take(p, 65536 * 2);
    _Float16* qvh1  = (_Float16*)take(p, 524288 * 2);
    _Float16* qvh2  = (_Float16*)take(p, 524288 * 2);
    _Float16* v1pk  = (_Float16*)take(p, 262144 * 2);   // packed v fragments
    _Float16* v2pk  = (_Float16*)take(p, 262144 * 2);
    float*    dotb  = (float*)take(p, 2097152 * 4);
    float*    ms1   = (float*)take(p, 2097152 * 4);
    float*    ms2   = (float*)take(p, 2097152 * 4);
    _Float16* ho1h  = (_Float16*)take(p, 262144 * 2);
    _Float16* ho2h  = (_Float16*)take(p, 262144 * 2);
    float*    accb  = (float*)take(p, 8 * 4);

    // A-side f32 -> f16 (row-major), B-side pack+convert (fragment-major)
    k_f2h<<<(262144 + 255) / 256, 256, 0, stream>>>(x1, x1h, 262144);
    k_f2h<<<(262144 + 255) / 256, 256, 0, stream>>>(x2, x2h, 262144);
    k_pack_b<<<64, 128, 0, stream>>>(Wqv1, Wqp, 256, 512);   // 256 frags
    k_pack_b<<< 8, 128, 0, stream>>>(lin1, l1p,  32, 512);   // 32 frags
    k_pack_b<<< 8, 128, 0, stream>>>(lin2, l2p, 512,  32);   // 32 frags
    k_pack_b<<<32, 128, 0, stream>>>(out1, o1p, 256, 256);   // 128 frags
    k_pack_b<<<32, 128, 0, stream>>>(out2, o2p, 256, 256);
    k_zero<<<1, 32, 0, stream>>>(accb, 8);

    // qv = x @ Wqv1   (M=1024, N=512, K=256) -> f16
    k_gemm_f16<false><<<512, 128, 0, stream>>>(x1h, Wqp, nullptr, qvh1, 1024, 512, 256);
    k_gemm_f16<false><<<512, 128, 0, stream>>>(x2h, Wqp, nullptr, qvh2, 1024, 512, 256);

    // pack v-parts for attention combine (512 waves each)
    k_packv<<<128, 128, 0, stream>>>(qvh1, v1pk);
    k_packv<<<128, 128, 0, stream>>>(qvh2, v2pk);

    // dot = 0.25 * q@k^T   (8192 tiles, 4 waves/block)
    k_dot<<<2048, 128, 0, stream>>>(qvh1, qvh2, dotb);

    // fused MLP -> ms1, ms2 + stats   (8192 waves, 2 waves/block, async lin2 staging)
    k_mlp<<<4096, 64, 0, stream>>>(dotb, cost, l1p, l2p, ms1, ms2, accb);

    // scalar unbiased std -> inverse
    k_std<<<1, 1, 0, stream>>>(accb);

    // attention combine: h1 uses ms1 & v2; h2 uses ms2^T & v1   (1024 waves each)
    k_att<<<256, 128, 0, stream>>>(ms1, v2pk, accb, 4, 0, ho1h);
    k_att<<<256, 128, 0, stream>>>(ms2, v1pk, accb, 5, 1, ho2h);

    // output projections -> d_out (f32)
    k_gemm_f16<true><<<256, 128, 0, stream>>>(ho1h, o1p, dOut,          nullptr, 1024, 256, 256);
    k_gemm_f16<true><<<256, 128, 0, stream>>>(ho2h, o2p, dOut + 262144, nullptr, 1024, 256, 256);
}
